// DAFCN_87840671137969
// MI455X (gfx1250) — matrix operand order repaired and checked by reading, not compiled, not run on hardware
//
#include <hip/hip_runtime.h>
#include <hip/hip_bf16.h>
#include <math.h>

// ---------------------------------------------------------------------------
// DAFCN forward for MI455X (gfx1250).
// The attention branch (wq/wk convs, dvb, attn) is dead code in the reference
// (combined[:, :, :10] == gcn_out), so only GCN + FFC + MLP survive.
// Heavy compute: per batch item 4x (48x512)@(512x512) GEMMs and 5x
// (48x48)@(48x512) att mixes -> both on v_wmma_f32_16x16x32_f16 (f16 in,
// f32 acc). GEMM1 stores its D-tiles transposed in f16 so they are directly
// consumable as B-fragments by the att WMMA. tanh is a branch-free
// exp/rcp-based approximation (error << f16 activation quantization).
// ---------------------------------------------------------------------------

typedef __attribute__((ext_vector_type(16))) _Float16 v16h;
typedef __attribute__((ext_vector_type(8)))  _Float16 v8h;
typedef __attribute__((ext_vector_type(8)))  float    v8f;

#define BATCH   1024
#define FEAT    48
#define DMODEL  512
#define BLEN    30      // KS + output_n
#define DCTN    10
#define NTILE   32      // 512/16
#define KSTEPS  16      // 512/32
#define LAYERHALVES (NTILE * KSTEPS * 32 * 16)   // 262144 halves per 512x512 layer
#define ATTSTAGEHALVES (3 * 2 * 32 * 16)         // 3072 halves per packed att

// dctm[d][k] = w_d * cos(pi*(k+0.5)*d/30); orthonormal -> inv(dctm) = dctm^T
__device__ __forceinline__ float dctc(int d, int k) {
    float w = (d == 0) ? 0.18257418583505536f   // sqrt(1/30)
                       : 0.25819888974716115f;  // sqrt(2/30)
    return w * cosf(0.10471975511965977f * ((float)k + 0.5f) * (float)d);
}

// Branch-free tanh: sign(x) * (1-e)/(1+e), e = exp(-2|x|).
// Uses v_exp_f32 + v_rcp_f32; |err| ~1e-7, well below f16 storage rounding.
__device__ __forceinline__ float fast_tanh(float x) {
    float ax = __builtin_fabsf(x);
    float e  = __expf(-2.0f * ax);
    float t  = (1.0f - e) * __builtin_amdgcn_rcpf(1.0f + e);
    return __builtin_copysignf(t, x);
}

// ---------------------------------------------------------------------------
// Pack one 512x512 f32 weight matrix (w[k][n], k = input dim) into WMMA
// B-fragment order: wpack[((L*32+nt)*16+ks)*512 + lane*16 + e] where
//   n = nt*16 + lane%16,  k = ks*32 + (lane/16)*16 + e
// ---------------------------------------------------------------------------
__global__ void pack_weights(const float* __restrict__ gcb_w,
                             _Float16* __restrict__ wpack) {
    int blk  = blockIdx.x;          // 0..2047 = L*512 + nt*16 + ks
    int L    = blk >> 9;
    int rem  = blk & 511;
    int nt   = rem >> 4;
    int ks   = rem & 15;
    int lane = threadIdx.x;         // 0..31
    int n     = nt * 16 + (lane & 15);
    int kbase = ks * 32 + (lane >> 4) * 16;
    const float* src = gcb_w + (size_t)L * DMODEL * DMODEL;
    _Float16* dst = wpack + (size_t)blk * 512 + (size_t)lane * 16;
#pragma unroll
    for (int e = 0; e < 16; ++e)
        dst[e] = (_Float16)src[(size_t)(kbase + e) * DMODEL + n];
}

// ---------------------------------------------------------------------------
// Pack the 5 att matrices (gc1, 4x gcb) into WMMA A-fragment order with K
// zero-padded 48 -> 64:  attpack[(((A*3+mt)*2+ks)*32+lane)*16 + e] where
//   rowM = mt*16 + lane%16,  k = ks*32 + g*16 + e
// ---------------------------------------------------------------------------
__global__ void pack_att(const float* __restrict__ gc1_att,
                         const float* __restrict__ gcb_att,
                         _Float16* __restrict__ attpack) {
    int blk  = blockIdx.x;          // 0..29 = (A*3+mt)*2+ks
    int A    = blk / 6;
    int mt   = (blk % 6) >> 1;
    int ks   = blk & 1;
    int lane = threadIdx.x;
    int g    = lane >> 4;
    const float* src = (A == 0) ? gc1_att
                                : (gcb_att + (size_t)(A - 1) * FEAT * FEAT);
    int n = mt * 16 + (lane & 15);
    _Float16* dst = attpack + ((size_t)blk * 32 + lane) * 16;
#pragma unroll
    for (int e = 0; e < 16; ++e) {
        int k = ks * 32 + g * 16 + e;
        dst[e] = (_Float16)((k < FEAT) ? src[n * FEAT + k] : 0.f);
    }
}

// ---------------------------------------------------------------------------
// FFC branch: exact rfft(60) as DFT matmul, wg channel mix + relu, partial
// irfft for the first 10 time steps, plus the wl local term.
// ffc_ws layout: [b][t<10][f=48]
// ---------------------------------------------------------------------------
__global__ void __launch_bounds__(256) ffc_kernel(const float* __restrict__ seq,
                                                  const float* __restrict__ wl,
                                                  const float* __restrict__ wg,
                                                  float* __restrict__ ffc_ws) {
    __shared__ float xb[48 * 60];
    __shared__ float Rr[48 * 31];
    __shared__ float Ii[48 * 31];
    __shared__ float Zr[48 * 31];
    __shared__ float Zi[48 * 31];

    const int b = blockIdx.x, tid = threadIdx.x;
    const float W = 6.283185307179586f / 60.0f;

    for (int i = tid; i < 48 * 60; i += 256) {
        int cf = i / 60, t = i % 60;
        int ts = t < 50 ? t : 49;                 // repeat last row
        xb[i] = seq[((size_t)b * 50 + ts) * FEAT + cf];
    }
    __syncthreads();

    for (int i = tid; i < 48 * 31; i += 256) {
        int cf = i / 31, k = i % 31;
        float sr = 0.f, si = 0.f;
        for (int t = 0; t < 60; ++t) {
            float sn, cs;
            sincosf(W * (float)(k * t), &sn, &cs);
            float xv = xb[cf * 60 + t];
            sr += xv * cs;
            si -= xv * sn;
        }
        Rr[i] = sr; Ii[i] = si;
    }
    __syncthreads();

    for (int i = tid; i < 16 * 31; i += 256) {
        int fg = i / 31, k = i % 31;
        float yc[6];
#pragma unroll
        for (int c = 0; c < 3; ++c) {
            yc[c]     = Rr[(c * 16 + fg) * 31 + k];
            yc[3 + c] = Ii[(c * 16 + fg) * 31 + k];
        }
#pragma unroll
        for (int o = 0; o < 3; ++o) {
            float zr = 0.f, zi = 0.f;
#pragma unroll
            for (int c = 0; c < 6; ++c) {
                zr += wg[o * 6 + c]       * yc[c];
                zi += wg[(o + 3) * 6 + c] * yc[c];
            }
            Zr[(o * 16 + fg) * 31 + k] = fmaxf(zr, 0.f);
            Zi[(o * 16 + fg) * 31 + k] = fmaxf(zi, 0.f);
        }
    }
    __syncthreads();

    for (int i = tid; i < 48 * 10; i += 256) {
        int cf = i / 10, t = i % 10;
        int c = cf / 16, fg = cf % 16;
        float s = Zr[cf * 31 + 0];
        float nyq = Zr[cf * 31 + 30];
        s += (t & 1) ? -nyq : nyq;
        for (int k = 1; k < 30; ++k) {
            float sn, cs;
            sincosf(W * (float)(k * t), &sn, &cs);
            s += 2.f * (Zr[cf * 31 + k] * cs - Zi[cf * 31 + k] * sn);
        }
        float spec = s * (1.0f / 60.0f);
        float loc = 0.f;
#pragma unroll
        for (int c2 = 0; c2 < 3; ++c2)
            loc += wl[c * 3 + c2] * xb[(c2 * 16 + fg) * 60 + t];
        ffc_ws[((size_t)b * 10 + t) * FEAT + cf] = loc + spec;
    }
}

// ---------------------------------------------------------------------------
// GEMM1: (48x512) = A(48x512 f16, LDS row-major) @ W(512x512 f16-packed).
// D-tiles are stored TRANSPOSED in f16: tmpT[n*64 + m]  (one ds_store_b128),
// which is exactly the B-fragment layout the att WMMA consumes.
// ---------------------------------------------------------------------------
__device__ __forceinline__ void gemm48x512x512(const _Float16* A,
                                               const _Float16* __restrict__ Bpack,
                                               _Float16* tmpT, int tid) {
    const int lane = tid & 31, wave = tid >> 5;
    const int g = lane >> 4, r = lane & 15;
    for (int i = 0; i < 12; ++i) {
        const int t  = wave + (i << 3);   // 0..95
        const int mt = t >> 5;            // 0..2
        const int nt = t & 31;            // 0..31
        v8f acc = {};
        const _Float16* arow  = A + (mt * 16 + r) * DMODEL;
        const _Float16* bbase = Bpack + (size_t)nt * (KSTEPS * 32 * 16)
                                      + (size_t)lane * 16;
#pragma unroll
        for (int ks = 0; ks < KSTEPS; ++ks) {
            v8h lo = *(const v8h*)(arow + ks * 32 + g * 16);
            v8h hi = *(const v8h*)(arow + ks * 32 + g * 16 + 8);
            union { v16h v; v8h h[2]; } ua;
            ua.h[0] = lo; ua.h[1] = hi;
            v16h bf = *(const v16h*)(bbase + (size_t)ks * (32 * 16));
            acc = __builtin_amdgcn_wmma_f32_16x16x32_f16(
                      false, ua.v, false, bf, (short)0, acc, false, false);
        }
        // D: m = mt*16 + g*8 + v (contiguous in v), n = nt*16 + r
        v8h hv;
#pragma unroll
        for (int v = 0; v < 8; ++v) hv[v] = (_Float16)acc[v];
        *(v8h*)(tmpT + (nt * 16 + r) * 64 + mt * 16 + g * 8) = hv;
    }
}

// ---------------------------------------------------------------------------
// att WMMA: dst[n][o] = tanh(bias[o] + sum_m att[n][m] * tmpT[o][m]),
// i.e. C(48x512) = attpack(48x64, A-frag, global) @ tmpT(64x512, B-frag, LDS).
// Bias + tanh applied on accumulator registers, f16 store to activation panel.
// ---------------------------------------------------------------------------
__device__ __forceinline__ void att_wmma_tanh(const _Float16* __restrict__ attp,
                                              const _Float16* tmpT,
                                              const float* __restrict__ bias,
                                              _Float16* dst, int tid) {
    const int lane = tid & 31, wave = tid >> 5;
    const int g = lane >> 4, r = lane & 15;
    for (int i = 0; i < 12; ++i) {
        const int t  = wave + (i << 3);
        const int mt = t >> 5, nt = t & 31;
        v8f acc = {};
#pragma unroll
        for (int ks = 0; ks < 2; ++ks) {
            v16h a  = *(const v16h*)(attp + (((mt * 2 + ks) * 32) + lane) * 16);
            v16h bf = *(const v16h*)(tmpT + (nt * 16 + r) * 64 + ks * 32 + g * 16);
            acc = __builtin_amdgcn_wmma_f32_16x16x32_f16(
                      false, a, false, bf, (short)0, acc, false, false);
        }
        const float bv = bias[nt * 16 + r];
        _Float16* d0 = dst + (mt * 16 + g * 8) * DMODEL + nt * 16 + r;
#pragma unroll
        for (int v = 0; v < 8; ++v)
            d0[v * DMODEL] = (_Float16)fast_tanh(acc[v] + bv);
    }
}

// ---------------------------------------------------------------------------
// GCN + DCT/IDCT + MLP tail. One workgroup (8 wave32) per batch item.
// Dynamic LDS (174976 B): tmpT f16 (64KB, reused as f32 scratch in the tail)
// | att f32 | dct_in f32 | aY f16 | aH f16
// ---------------------------------------------------------------------------
__global__ void __launch_bounds__(256) gcn_kernel(
    const float* __restrict__ seq,
    const float* __restrict__ gc1_w, const float* __restrict__ gc1_b,
    const float* __restrict__ gcb_b,
    const float* __restrict__ gc7_w,  const float* __restrict__ gc7_att,
    const float* __restrict__ gc7_b,
    const float* __restrict__ mlp_w1, const float* __restrict__ mlp_w2,
    const _Float16* __restrict__ wpack, const _Float16* __restrict__ attpack,
    const float* __restrict__ ffc_ws,
    float* __restrict__ out) {
    const int b = blockIdx.x, tid = threadIdx.x;

    extern __shared__ char smem_raw[];
    _Float16* tmpT  = (_Float16*)smem_raw;             // 32768 halves (65536 B)
    float*    att   = (float*)(smem_raw + 65536);      // 2304 f32
    float*    dctin = att + 2304;                      // 480 f32
    _Float16* aY    = (_Float16*)(dctin + 480);        // 24576 halves
    _Float16* aH    = aY + 24576;                      // 24576 halves

    // ---- zero-fill tmpT K-padding rows (k = 48..63) once
    for (int i = tid; i < DMODEL * 16; i += 256) {
        int n = i >> 4, j = i & 15;
        tmpT[n * 64 + 48 + j] = (_Float16)0.f;
    }

    // ---- P0: gather gcn_in rows (gidx = 40..49, then 49 x20), DCT -> 10 coeffs
    float* seqbuf = att;                               // 1440 <= 2304
    for (int i = tid; i < BLEN * FEAT; i += 256) {
        int k = i / FEAT, f = i % FEAT;
        int row = (k < 10) ? (40 + k) : 49;
        seqbuf[i] = seq[((size_t)b * 50 + row) * FEAT + f];
    }
    __syncthreads();
    for (int idx = tid; idx < FEAT * DCTN; idx += 256) {
        int m = idx / DCTN, d = idx % DCTN;
        float s = 0.f;
        for (int k = 0; k < BLEN; ++k) s += dctc(d, k) * seqbuf[k * FEAT + m];
        dctin[idx] = s;
    }
    __syncthreads();

    // ---- gc1: tmpT = (dct_in @ gc1_w)^T  (K=10, VALU -> f16 transposed store)
    for (int idx = tid; idx < FEAT * DMODEL; idx += 256) {
        int m = idx >> 9, o = idx & 511;
        float s = 0.f;
#pragma unroll
        for (int d = 0; d < DCTN; ++d) s += dctin[m * DCTN + d] * gc1_w[d * DMODEL + o];
        tmpT[o * 64 + m] = (_Float16)s;
    }
    __syncthreads();
    att_wmma_tanh(attpack, tmpT, gc1_b, aY, tid);      // stage A=0
    __syncthreads();

    // ---- residual blocks: h = tanh(att @ (x @ W) + b) twice; y += h
    for (int s2 = 0; s2 < 2; ++s2) {
        for (int half = 0; half < 2; ++half) {
            const int L = s2 * 2 + half;
            const _Float16* Ain = (half == 0) ? aY : aH;
            gemm48x512x512(Ain, wpack + (size_t)L * LAYERHALVES, tmpT, tid);
            __syncthreads();
            att_wmma_tanh(attpack + (size_t)(L + 1) * ATTSTAGEHALVES, tmpT,
                          gcb_b + (size_t)L * DMODEL, aH, tid);
            __syncthreads();
        }
        for (int i = tid; i < FEAT * DMODEL; i += 256)
            aY[i] = (_Float16)((float)aY[i] + (float)aH[i]);
        __syncthreads();
    }

    // ---- gc7 (512 -> 10) + att + bias + skip(dct_in)  (VALU, tiny)
    float* fsc  = (float*)smem_raw;   // reuse tmpT area as 16384 f32 scratch
    float* t7   = fsc;                // 480
    float* gout = fsc + 512;          // 480
    for (int i = tid; i < FEAT * FEAT; i += 256) att[i] = gc7_att[i];
    for (int idx = tid; idx < FEAT * DCTN; idx += 256) {
        int m = idx / DCTN, j = idx % DCTN;
        float s = 0.f;
        for (int k = 0; k < DMODEL; ++k)
            s += (float)aY[m * DMODEL + k] * gc7_w[k * DCTN + j];
        t7[idx] = s;
    }
    __syncthreads();
    for (int idx = tid; idx < FEAT * DCTN; idx += 256) {
        int n = idx / DCTN, j = idx % DCTN;
        float s = gc7_b[j] + dctin[idx];
#pragma unroll 8
        for (int m = 0; m < FEAT; ++m) s += att[n * FEAT + m] * t7[m * DCTN + j];
        gout[idx] = s;
    }
    __syncthreads();

    // ---- fused[f][t] : t<30 = IDCT(gcn_out) (idctm = dctm^T), t>=30 = ffc_out
    float* fused = fsc + 1024;  // 1920
    float* mlph  = fsc + 3072;  // 12288
    for (int idx = tid; idx < FEAT * 40; idx += 256) {
        int f = idx / 40, t = idx % 40;
        float v;
        if (t < 30) {
            v = 0.f;
#pragma unroll
            for (int d = 0; d < DCTN; ++d) v += dctc(d, t) * gout[f * DCTN + d];
        } else {
            v = ffc_ws[((size_t)b * 10 + (t - 30)) * FEAT + f];
        }
        fused[idx] = v;
    }
    __syncthreads();

    // ---- MLP tail: h = relu(fused @ w1^T); out = h @ w2^T; keep t<10
    for (int idx = tid; idx < FEAT * 256; idx += 256) {
        int f = idx >> 8, o = idx & 255;
        float s = 0.f;
#pragma unroll 8
        for (int t = 0; t < 40; ++t) s += fused[f * 40 + t] * mlp_w1[o * 40 + t];
        mlph[idx] = fmaxf(s, 0.f);
    }
    __syncthreads();
    for (int idx = tid; idx < FEAT * DCTN; idx += 256) {
        int f = idx / DCTN, t = idx % DCTN;
        float s = 0.f;
#pragma unroll 8
        for (int o = 0; o < 256; ++o) s += mlph[f * 256 + o] * mlp_w2[t * 256 + o];
        out[((size_t)b * DCTN + t) * FEAT + f] = s;   // (B, 10, 1, 48)
    }
}

// ---------------------------------------------------------------------------
extern "C" void kernel_launch(void* const* d_in, const int* in_sizes, int n_in,
                              void* d_out, int out_size, void* d_ws, size_t ws_size,
                              hipStream_t stream) {
    (void)in_sizes; (void)n_in; (void)out_size; (void)ws_size;
    const float* seq     = (const float*)d_in[0];
    const float* gc1_w   = (const float*)d_in[5];
    const float* gc1_att = (const float*)d_in[6];
    const float* gc1_b   = (const float*)d_in[7];
    const float* gcb_w   = (const float*)d_in[8];
    const float* gcb_att = (const float*)d_in[9];
    const float* gcb_b   = (const float*)d_in[10];
    const float* gc7_w   = (const float*)d_in[11];
    const float* gc7_att = (const float*)d_in[12];
    const float* gc7_b   = (const float*)d_in[13];
    const float* mlp_w1  = (const float*)d_in[14];
    const float* mlp_w2  = (const float*)d_in[15];
    const float* ffc_wl  = (const float*)d_in[16];
    const float* ffc_wg  = (const float*)d_in[17];
    float* out = (float*)d_out;

    char* ws = (char*)d_ws;
    float*    ffc_ws  = (float*)ws;                     // 1,966,080 B
    _Float16* wpack   = (_Float16*)(ws + 1966080);      // 2 MiB
    _Float16* attpack = (_Float16*)(ws + 1966080 + 2097152);  // 30,720 B

    const int smem = 65536 + (2304 + 480) * 4 + 2 * 24576 * 2;  // 174976 B
    hipFuncSetAttribute((const void*)gcn_kernel,
                        hipFuncAttributeMaxDynamicSharedMemorySize, smem);

    pack_weights<<<2048, 32, 0, stream>>>(gcb_w, wpack);
    pack_att<<<30, 32, 0, stream>>>(gc1_att, gcb_att, attpack);
    ffc_kernel<<<BATCH, 256, 0, stream>>>(seq, ffc_wl, ffc_wg, ffc_ws);
    gcn_kernel<<<BATCH, 256, smem, stream>>>(seq, gc1_w, gc1_b,
                                             gcb_b,
                                             gc7_w, gc7_att, gc7_b,
                                             mlp_w1, mlp_w2,
                                             wpack, attpack, ffc_ws, out);
}
